// TreatmentGNN_15917148799744
// MI455X (gfx1250) — compile-verified, matrix operand order checked by reading, not run on hardware
//
#include <hip/hip_runtime.h>

typedef __attribute__((ext_vector_type(16))) _Float16 v16h;
typedef __attribute__((ext_vector_type(8)))  float    v8f;

#define NPROV 100000
#define NCODE 20000
#define NEDGE 2000000
#define DIM   128
#define HID   128
#define OUTD  64

// ---------------------------------------------------------------------------
// Pack a row-major f32 weight matrix [K x N] into f16 WMMA B-fragments.
// Fragment (c,t) covers K-chunk 32c..32c+31, N-tile 16t..16t+15.
// Per lane l: column n = 16t + (l&15); 16 K values in two runs of 8, first at
// 32c + 8*(l>>4) ... split as e<8 -> k=32c+8*(l>>4)+e ; e>=8 -> +16.
// Linear index: ((c*NT + t)*32 + lane)*16 + e
// ---------------------------------------------------------------------------
__global__ __launch_bounds__(256) void pack_w(const float* __restrict__ w,
                                              _Float16* __restrict__ pk,
                                              int K, int N) {
  int idx = blockIdx.x * 256 + threadIdx.x;
  if (idx >= K * N) return;
  int e    = idx & 15;
  int lane = (idx >> 4) & 31;
  int frag = idx >> 9;
  int NT   = N >> 4;
  int t    = frag % NT;
  int c    = frag / NT;
  int hlf  = lane >> 4;
  int col  = lane & 15;
  int k = 32 * c + ((e < 8) ? (8 * hlf + e) : (16 + 8 * hlf + (e - 8)));
  int n = 16 * t + col;
  pk[idx] = (_Float16)w[k * N + n];
}

// ---------------------------------------------------------------------------
// One wave32 per edge: gather 128-float neighbor embedding (float4 per lane),
// scale by edge weight, scatter-add with native f32 global atomics. Lane 0
// accumulates weighted degrees.
// ---------------------------------------------------------------------------
__global__ __launch_bounds__(256) void edge_scatter(
    const int* __restrict__ eprov, const int* __restrict__ ecode,
    const float* __restrict__ ew,
    const float* __restrict__ pemb, const float* __restrict__ cemb,
    float* __restrict__ p_acc, float* __restrict__ c_acc,
    float* __restrict__ p_deg, float* __restrict__ c_deg, int nedges) {
  int gid  = blockIdx.x * blockDim.x + threadIdx.x;
  int edge = gid >> 5;
  int lane = gid & 31;
  if (edge >= nedges) return;
  int   p = eprov[edge];
  int   c = ecode[edge];
  float w = ew[edge];

  const float4* sc = (const float4*)(cemb + (size_t)c * DIM);
  float4 vc = sc[lane];
  float* dp = p_acc + (size_t)p * DIM + lane * 4;
  unsafeAtomicAdd(dp + 0, vc.x * w);
  unsafeAtomicAdd(dp + 1, vc.y * w);
  unsafeAtomicAdd(dp + 2, vc.z * w);
  unsafeAtomicAdd(dp + 3, vc.w * w);

  const float4* sp = (const float4*)(pemb + (size_t)p * DIM);
  float4 vp = sp[lane];
  float* dc = c_acc + (size_t)c * DIM + lane * 4;
  unsafeAtomicAdd(dc + 0, vp.x * w);
  unsafeAtomicAdd(dc + 1, vp.y * w);
  unsafeAtomicAdd(dc + 2, vp.z * w);
  unsafeAtomicAdd(dc + 3, vp.w * w);

  if (lane == 0) {
    unsafeAtomicAdd(p_deg + p, w);
    unsafeAtomicAdd(c_deg + c, w);
  }
}

// ---------------------------------------------------------------------------
// One wave32 per 16-row tile. X = [emb | acc/deg] (16x256 f32 -> f16).
// Layer1: 8 K-chunks x 8 N-tiles of v_wmma_f32_16x16x32_f16 -> 16x128 f32.
// Bias + LayerNorm (shfl_xor over 16-lane half) + ReLU -> LDS (f16, A-layout
// source). Layer2: 4x4 WMMA -> 16x64, bias + LayerNorm -> out (f32).
// ---------------------------------------------------------------------------
__global__ __launch_bounds__(32) void mlp16(
    const float* __restrict__ emb, const float* __restrict__ acc,
    const float* __restrict__ deg,
    const _Float16* __restrict__ pk1, const float* __restrict__ b1,
    const float* __restrict__ g1, const float* __restrict__ be1,
    const _Float16* __restrict__ pk2, const float* __restrict__ b2,
    const float* __restrict__ g2, const float* __restrict__ be2,
    float* __restrict__ out) {
  __shared__ _Float16 hbuf[16 * HID];

  const int lane = threadIdx.x;
  const int hlf  = lane >> 4;
  const int col  = lane & 15;
  const int rowA = blockIdx.x * 16 + col;       // A-matrix row owned by lane
  const float invdeg = 1.0f / (deg[rowA] + 1e-8f);

  const float* erow = emb + (size_t)rowA * DIM;
  const float* arow = acc + (size_t)rowA * DIM;

  v8f accs[8] = {};

  // ---- Layer 1: X[16x256] @ W1[256x128] ----
#pragma unroll
  for (int c = 0; c < 8; ++c) {
    const float* src = (c < 4) ? erow : arow;
    const float scale = (c < 4) ? 1.0f : invdeg;
    const int kb = 32 * (c & 3);
    v16h a;
#pragma unroll
    for (int e = 0; e < 8; ++e)
      a[e] = (_Float16)(src[kb + 8 * hlf + e] * scale);
#pragma unroll
    for (int e = 0; e < 8; ++e)
      a[8 + e] = (_Float16)(src[kb + 16 + 8 * hlf + e] * scale);
#pragma unroll
    for (int t = 0; t < 8; ++t) {
      v16h b = *(const v16h*)(pk1 + (((c * 8 + t) * 32 + lane) << 4));
      accs[t] = __builtin_amdgcn_wmma_f32_16x16x32_f16(
          false, a, false, b, (short)0, accs[t], false, false);
    }
  }

  // ---- bias + LayerNorm + ReLU -> LDS (f16) ----
  float b1v[8], g1v[8], be1v[8];
#pragma unroll
  for (int t = 0; t < 8; ++t) {
    int n = 16 * t + col;
    b1v[t] = b1[n]; g1v[t] = g1[n]; be1v[t] = be1[n];
  }
#pragma unroll
  for (int r = 0; r < 8; ++r) {
    float xv[8];
    float s = 0.0f, sq = 0.0f;
#pragma unroll
    for (int t = 0; t < 8; ++t) {
      xv[t] = accs[t][r] + b1v[t];
      s += xv[t];
      sq += xv[t] * xv[t];
    }
#pragma unroll
    for (int m = 1; m < 16; m <<= 1) {
      s  += __shfl_xor(s, m, 32);
      sq += __shfl_xor(sq, m, 32);
    }
    float mu  = s * (1.0f / 128.0f);
    float var = sq * (1.0f / 128.0f) - mu * mu;
    float rs  = rsqrtf(var + 1e-5f);
    int   mrow = r + 8 * hlf;
#pragma unroll
    for (int t = 0; t < 8; ++t) {
      float h = (xv[t] - mu) * rs * g1v[t] + be1v[t];
      h = fmaxf(h, 0.0f);
      hbuf[mrow * HID + 16 * t + col] = (_Float16)h;
    }
  }
  __syncthreads();

  // ---- Layer 2: H[16x128] @ W2[128x64] ----
  v8f acc2[4] = {};
#pragma unroll
  for (int c = 0; c < 4; ++c) {
    const _Float16* hrow = hbuf + col * HID + 32 * c;
    v16h a;
#pragma unroll
    for (int e = 0; e < 8; ++e) a[e] = hrow[8 * hlf + e];
#pragma unroll
    for (int e = 0; e < 8; ++e) a[8 + e] = hrow[16 + 8 * hlf + e];
#pragma unroll
    for (int t = 0; t < 4; ++t) {
      v16h b = *(const v16h*)(pk2 + (((c * 4 + t) * 32 + lane) << 4));
      acc2[t] = __builtin_amdgcn_wmma_f32_16x16x32_f16(
          false, a, false, b, (short)0, acc2[t], false, false);
    }
  }

  // ---- bias + LayerNorm -> out ----
  float b2v[4], g2v[4], be2v[4];
#pragma unroll
  for (int t = 0; t < 4; ++t) {
    int n = 16 * t + col;
    b2v[t] = b2[n]; g2v[t] = g2[n]; be2v[t] = be2[n];
  }
#pragma unroll
  for (int r = 0; r < 8; ++r) {
    float xv[4];
    float s = 0.0f, sq = 0.0f;
#pragma unroll
    for (int t = 0; t < 4; ++t) {
      xv[t] = acc2[t][r] + b2v[t];
      s += xv[t];
      sq += xv[t] * xv[t];
    }
#pragma unroll
    for (int m = 1; m < 16; m <<= 1) {
      s  += __shfl_xor(s, m, 32);
      sq += __shfl_xor(sq, m, 32);
    }
    float mu  = s * (1.0f / 64.0f);
    float var = sq * (1.0f / 64.0f) - mu * mu;
    float rs  = rsqrtf(var + 1e-5f);
    int   grow = blockIdx.x * 16 + r + 8 * hlf;
#pragma unroll
    for (int t = 0; t < 4; ++t)
      out[(size_t)grow * OUTD + 16 * t + col] =
          (xv[t] - mu) * rs * g2v[t] + be2v[t];
  }
}

// ---------------------------------------------------------------------------
extern "C" void kernel_launch(void* const* d_in, const int* in_sizes, int n_in,
                              void* d_out, int out_size, void* d_ws,
                              size_t ws_size, hipStream_t stream) {
  const int*   eprov = (const int*)d_in[0];
  const int*   ecode = (const int*)d_in[1];
  const float* ew    = (const float*)d_in[2];
  const float* pemb  = (const float*)d_in[3];
  const float* cemb  = (const float*)d_in[4];
  const float* w1p = (const float*)d_in[5];
  const float* b1p = (const float*)d_in[6];
  const float* g1p = (const float*)d_in[7];
  const float* be1p = (const float*)d_in[8];
  const float* w2p = (const float*)d_in[9];
  const float* b2p = (const float*)d_in[10];
  const float* g2p = (const float*)d_in[11];
  const float* be2p = (const float*)d_in[12];
  const float* w1c = (const float*)d_in[13];
  const float* b1c = (const float*)d_in[14];
  const float* g1c = (const float*)d_in[15];
  const float* be1c = (const float*)d_in[16];
  const float* w2c = (const float*)d_in[17];
  const float* b2c = (const float*)d_in[18];
  const float* g2c = (const float*)d_in[19];
  const float* be2c = (const float*)d_in[20];

  char* ws = (char*)d_ws;
  // workspace layout (bytes)
  float*    p_acc = (float*)(ws + 0);                 // 100000*128*4 = 51,200,000
  float*    c_acc = (float*)(ws + 51200000);          //  20000*128*4 = 10,240,000
  float*    p_deg = (float*)(ws + 61440000);          // 100000*4     =    400,000
  float*    c_deg = (float*)(ws + 61840000);          //  20000*4     =     80,000
  _Float16* pk1p  = (_Float16*)(ws + 61920000);       // 256*128*2    =     65,536
  _Float16* pk2p  = (_Float16*)(ws + 61985536);       // 128*64*2     =     16,384
  _Float16* pk1c  = (_Float16*)(ws + 62001920);       //                    65,536
  _Float16* pk2c  = (_Float16*)(ws + 62067456);       //                    16,384

  // zero accumulators + degrees (graph-capturable memset)
  hipMemsetAsync(d_ws, 0, 61920000, stream);

  // pack weights into f16 B-fragments
  pack_w<<<(256 * 128 + 255) / 256, 256, 0, stream>>>(w1p, pk1p, 256, 128);
  pack_w<<<(128 * 64 + 255) / 256, 256, 0, stream>>>(w2p, pk2p, 128, 64);
  pack_w<<<(256 * 128 + 255) / 256, 256, 0, stream>>>(w1c, pk1c, 256, 128);
  pack_w<<<(128 * 64 + 255) / 256, 256, 0, stream>>>(w2c, pk2c, 128, 64);

  // message passing: one wave32 per edge, 8 waves per 256-thread block
  int eblocks = (NEDGE + 7) / 8;
  edge_scatter<<<eblocks, 256, 0, stream>>>(eprov, ecode, ew, pemb, cemb,
                                            p_acc, c_acc, p_deg, c_deg, NEDGE);

  // MLPs: one wave32 per 16-row tile
  float* out_p = (float*)d_out;
  float* out_c = out_p + (size_t)NPROV * OUTD;
  mlp16<<<NPROV / 16, 32, 0, stream>>>(pemb, p_acc, p_deg,
                                       pk1p, b1p, g1p, be1p,
                                       pk2p, b2p, g2p, be2p, out_p);
  mlp16<<<NCODE / 16, 32, 0, stream>>>(cemb, c_acc, c_deg,
                                       pk1c, b1c, g1c, be1c,
                                       pk2c, b2c, g2c, be2c, out_c);
}
